// CascadeNNBN_9345848836169
// MI455X (gfx1250) — matrix operand order repaired and checked by reading, not compiled, not run on hardware
//
#include <hip/hip_runtime.h>
#include <hip/hip_bf16.h>

// ---------------------------------------------------------------------------
// CascadeNN + BatchNorm on MI455X (gfx1250), wave32.
// bf16 WMMA (f32 accum) + double-buffered LDS staging fed by
// GLOBAL_LOAD_ASYNC_TO_LDS_B128 (ASYNCcnt) for the CDNA5 async copy path.
// ---------------------------------------------------------------------------

#define B_BATCH  16384
#define D_IN     512
#define K_STAGES 8
#define W_STAGE  256
#define D_OUT    128
#define D_FINAL  2560          // 512 + 8*256
#define BN_EPS   1e-5f

#define LDS_PITCH 40           // 32 data elems + 8 pad (80 B row, 16B aligned,
                               // 20-bank row stride -> conflict-free frag reads)

typedef __attribute__((ext_vector_type(16))) __bf16        bf16x16;
typedef __attribute__((ext_vector_type(8)))  float         f32x8;
typedef __attribute__((ext_vector_type(4)))  unsigned int  u32x4;

union Frag { bf16x16 v; u32x4 q[2]; };   // 32 bytes = 16 bf16

// ---------------------------------------------------------------------------
// Conversion kernels
// ---------------------------------------------------------------------------
__global__ void cvt_x_kernel(__hip_bfloat16* __restrict__ feats,
                             const float* __restrict__ x) {
    int idx = blockIdx.x * blockDim.x + threadIdx.x;          // [B*512)
    if (idx >= B_BATCH * D_IN) return;
    int m = idx >> 9, n = idx & 511;
    feats[(size_t)m * D_FINAL + n] = __float2bfloat16(x[idx]);
}

__global__ void cvt_flat_kernel(__hip_bfloat16* __restrict__ dst,
                                const float* __restrict__ src, int n) {
    int idx = blockIdx.x * blockDim.x + threadIdx.x;
    if (idx < n) dst[idx] = __float2bfloat16(src[idx]);
}

// ---------------------------------------------------------------------------
// Stats helpers
// ---------------------------------------------------------------------------
__global__ void zero_stats_kernel(float* __restrict__ stats) {
    stats[threadIdx.x] = 0.0f;                                 // 512 threads
}

__global__ void bn_finalize_kernel(const float* __restrict__ colsum,
                                   const float* __restrict__ colsumsq,
                                   const float* __restrict__ gamma,
                                   const float* __restrict__ beta,
                                   float* __restrict__ aOut,
                                   float* __restrict__ cOut) {
    int n = threadIdx.x;                                       // 256 threads
    const float invB = 1.0f / (float)B_BATCH;
    float mu  = colsum[n] * invB;
    float var = colsumsq[n] * invB - mu * mu;
    float a   = gamma[n] * rsqrtf(var + BN_EPS);
    aOut[n] = a;
    cOut[n] = beta[n] - mu * a;
}

__global__ void bn_apply_kernel(__hip_bfloat16* __restrict__ slice, // feats + colOff
                                const float* __restrict__ a,
                                const float* __restrict__ c) {
    int idx = blockIdx.x * blockDim.x + threadIdx.x;           // [B*256)
    if (idx >= B_BATCH * W_STAGE) return;
    int m = idx >> 8, n = idx & 255;
    size_t p = (size_t)m * D_FINAL + n;
    float v = __bfloat162float(slice[p]);
    slice[p] = __float2bfloat16(a[n] * v + c[n]);
}

// ---------------------------------------------------------------------------
// Async global -> LDS copy of a 128-row x 32-col bf16 tile (row pitch LDS_PITCH).
// 512 x 16B chunks, 2 per thread. Tracked with ASYNCcnt.
// ---------------------------------------------------------------------------
__device__ __forceinline__ void async_copy_tile128x32(
        __hip_bfloat16* shDst, const __hip_bfloat16* __restrict__ gSrc,
        int ldg, int tid) {
#pragma unroll
    for (int t = 0; t < 2; ++t) {
        int chunk = tid * 2 + t;                 // 0..511
        int row   = chunk >> 2;                  // 0..127
        int c     = (chunk & 3) * 8;             // elem offset: 0,8,16,24
        unsigned int lds = (unsigned int)(uintptr_t)(shDst + row * LDS_PITCH + c);
        const __hip_bfloat16* g = gSrc + (size_t)row * ldg + c;
        asm volatile("global_load_async_to_lds_b128 %0, %1, off"
                     :: "v"(lds), "v"(g) : "memory");
    }
}

#define ASYNC_WAIT0() asm volatile("s_wait_asynccnt 0x0" ::: "memory")

// ---------------------------------------------------------------------------
// WMMA GEMM:  out[M,N] = A[M,K] (bf16, lda) @ Bw[N,K]^T (bf16, ldb) + bias[N]
// Block = 256 threads = 8 waves (4 along M x 2 along N).
// Block tile 128x128, wave tile 32x64 (8 WMMAs / 32-K step).
// Double-buffered LDS tiles filled by async global->LDS copies.
// MODE 0: ReLU, write bf16 slice (ldo = D_FINAL), atomic column sum/sumsq.
// MODE 1: no ReLU, write fp32 out (ld = D_OUT).
// ---------------------------------------------------------------------------
#define WMMA_BF16(a, b, c) \
    __builtin_amdgcn_wmma_f32_16x16x32_bf16(false, (a), false, (b), (short)0, (c), false, false)

template <int MODE>
__global__ __launch_bounds__(256) void wmma_gemm_kernel(
        const __hip_bfloat16* __restrict__ A, int lda,
        const __hip_bfloat16* __restrict__ Bw, int ldb, int Kdim,
        const float* __restrict__ bias,
        __hip_bfloat16* __restrict__ outB, int ldo,
        float* __restrict__ outF,
        float* __restrict__ colsum, float* __restrict__ colsumsq) {
    __shared__ __align__(16) __hip_bfloat16 shA[2][128 * LDS_PITCH];   // 2 x 10 KB
    __shared__ __align__(16) __hip_bfloat16 shB[2][128 * LDS_PITCH];   // 2 x 10 KB

    const int tid  = threadIdx.x;
    const int lane = tid & 31;
    const int half = lane >> 4;             // 0: lanes 0-15, 1: lanes 16-31
    const int l15  = lane & 15;
    const int wave = tid >> 5;
    const int wm   = wave & 3;              // 4 waves along M (32 rows each)
    const int wn   = wave >> 2;             // 2 waves along N (64 cols each)

    const size_t blkM0 = (size_t)blockIdx.x * 128;
    const int    blkN0 = blockIdx.y * 128;
    const int    wmBase = wm * 32;
    const int    wnBase = wn * 64;

    const __hip_bfloat16* gA = A  + blkM0 * (size_t)lda;
    const __hip_bfloat16* gB = Bw + (size_t)blkN0 * ldb;

    // Fragment sub-offsets within a 32-col LDS tile (bf16 A 16x32 / B 32x16 layouts).
    const int aoff0 = half * 16;
    const int aoff1 = half * 16 + 8;
    const int boff  = half * 16;

    f32x8 acc[2][4];
#pragma unroll
    for (int i = 0; i < 2; ++i)
#pragma unroll
        for (int j = 0; j < 4; ++j) acc[i][j] = (f32x8){};

    const int nk = Kdim >> 5;   // K steps of 32

    // Prologue: fill buffer 0.
    async_copy_tile128x32(shA[0], gA, lda, tid);
    async_copy_tile128x32(shB[0], gB, ldb, tid);

    for (int kt = 0; kt < nk; ++kt) {
        const int buf = kt & 1;
        ASYNC_WAIT0();          // own async copies for this buffer are done
        __syncthreads();        // everyone's copies visible; prev reads finished

        if (kt + 1 < nk) {      // overlap next tile copy with compute
            const int k0n = (kt + 1) << 5;
            async_copy_tile128x32(shA[buf ^ 1], gA + k0n, lda, tid);
            async_copy_tile128x32(shB[buf ^ 1], gB + k0n, ldb, tid);
        }

        Frag af[2];
#pragma unroll
        for (int i = 0; i < 2; ++i) {
            const __hip_bfloat16* pa = &shA[buf][(wmBase + i * 16 + l15) * LDS_PITCH];
            af[i].q[0] = *(const u32x4*)(pa + aoff0);
            af[i].q[1] = *(const u32x4*)(pa + aoff1);
        }
#pragma unroll
        for (int j = 0; j < 4; ++j) {
            Frag bf;
            const __hip_bfloat16* pb = &shB[buf][(wnBase + j * 16 + l15) * LDS_PITCH + boff];
            bf.q[0] = ((const u32x4*)pb)[0];
            bf.q[1] = ((const u32x4*)pb)[1];
            acc[0][j] = WMMA_BF16(af[0].v, bf.v, acc[0][j]);
            acc[1][j] = WMMA_BF16(af[1].v, bf.v, acc[1][j]);
        }
        __syncthreads();        // reads done before buffer is overwritten
    }

    // C/D layout: VGPR r -> M = r (lanes 0-15) or r+8 (lanes 16-31), N = lane&15.
#pragma unroll
    for (int j = 0; j < 4; ++j) {
        const int gn = blkN0 + wnBase + j * 16 + l15;
        const float bn = bias[gn];
        float s = 0.0f, ssq = 0.0f;
#pragma unroll
        for (int i = 0; i < 2; ++i) {
            const size_t gmBase = blkM0 + wmBase + i * 16 + half * 8;
            f32x8 av = acc[i][j];
#pragma unroll
            for (int r = 0; r < 8; ++r) {
                float h = av[r] + bn;
                if (MODE == 0) {
                    h = fmaxf(h, 0.0f);
                    outB[(gmBase + r) * (size_t)ldo + gn] = __float2bfloat16(h);
                    s += h;
                    ssq += h * h;
                } else {
                    outF[(gmBase + r) * (size_t)D_OUT + gn] = h;
                }
            }
        }
        if (MODE == 0) {
            // lane l and l+16 hold the same column; fold the two M-halves.
            s   += __shfl_xor(s, 16, 32);
            ssq += __shfl_xor(ssq, 16, 32);
            if (half == 0) {
                atomicAdd(&colsum[gn], s);
                atomicAdd(&colsumsq[gn], ssq);
            }
        }
    }
}

// ---------------------------------------------------------------------------
// Host launcher
// ---------------------------------------------------------------------------
extern "C" void kernel_launch(void* const* d_in, const int* in_sizes, int n_in,
                              void* d_out, int out_size, void* d_ws, size_t ws_size,
                              hipStream_t stream) {
    (void)in_sizes; (void)n_in; (void)out_size; (void)ws_size;

    const float* x     = (const float*)d_in[0];
    const float* W[K_STAGES];
    for (int i = 0; i < K_STAGES; ++i) W[i] = (const float*)d_in[1 + i];
    const float* bvec  = (const float*)d_in[9];    // [K,256]
    const float* gamma = (const float*)d_in[10];   // [K,256]
    const float* beta  = (const float*)d_in[11];   // [K,256]
    const float* Wout  = (const float*)d_in[12];   // [128,2560]
    const float* bout  = (const float*)d_in[13];   // [128]
    float* out = (float*)d_out;

    // Workspace layout (bf16 storage, 16B-aligned by construction):
    __hip_bfloat16* featsB = (__hip_bfloat16*)d_ws;               // [B,2560] bf16
    __hip_bfloat16* Wb     = featsB + (size_t)B_BATCH * D_FINAL;  // 256*11264 elems
    const size_t wbElems   = (size_t)256 * 11264;
    __hip_bfloat16* WoutB  = Wb + wbElems;                        // 128*2560 elems
    float* stats  = (float*)(WoutB + (size_t)D_OUT * D_FINAL);
    float* colsum   = stats;        // 256
    float* colsumsq = stats + 256;  // 256
    float* aArr     = stats + 512;  // 256
    float* cArr     = stats + 768;  // 256

    // 1) Convert inputs/weights to bf16.
    {
        int n = B_BATCH * D_IN;
        cvt_x_kernel<<<(n + 255) / 256, 256, 0, stream>>>(featsB, x);
    }
    size_t cum = 0;
    for (int i = 0; i < K_STAGES; ++i) {
        int d = D_IN + W_STAGE * i;
        int n = W_STAGE * d;
        cvt_flat_kernel<<<(n + 255) / 256, 256, 0, stream>>>(Wb + cum, W[i], n);
        cum += (size_t)n;
    }
    {
        int n = D_OUT * D_FINAL;
        cvt_flat_kernel<<<(n + 255) / 256, 256, 0, stream>>>(WoutB, Wout, n);
    }

    // 2) Cascade stages.
    cum = 0;
    for (int i = 0; i < K_STAGES; ++i) {
        const int Kdim = D_IN + W_STAGE * i;      // 512,768,...,2304
        zero_stats_kernel<<<1, 512, 0, stream>>>(stats);

        dim3 grid(B_BATCH / 128, W_STAGE / 128);  // (128, 2)
        wmma_gemm_kernel<0><<<grid, 256, 0, stream>>>(
            featsB, D_FINAL,
            Wb + cum, Kdim, Kdim,
            bvec + i * W_STAGE,
            featsB + Kdim /*column offset into feats*/, D_FINAL,
            nullptr,
            colsum, colsumsq);

        bn_finalize_kernel<<<1, 256, 0, stream>>>(
            colsum, colsumsq, gamma + i * W_STAGE, beta + i * W_STAGE, aArr, cArr);

        int n = B_BATCH * W_STAGE;
        bn_apply_kernel<<<(n + 255) / 256, 256, 0, stream>>>(featsB + Kdim, aArr, cArr);

        cum += (size_t)W_STAGE * Kdim;
    }

    // 3) Output GEMM -> fp32.
    {
        dim3 grid(B_BATCH / 128, D_OUT / 128);    // (128, 1)
        wmma_gemm_kernel<1><<<grid, 256, 0, stream>>>(
            featsB, D_FINAL,
            WoutB, D_FINAL, D_FINAL,
            bout,
            nullptr, 0,
            out,
            nullptr, nullptr);
    }
}
